// LancetBlock_23227183136969
// MI455X (gfx1250) — matrix operand order repaired and checked by reading, not compiled, not run on hardware
//
#include <hip/hip_runtime.h>
#include <hip/hip_bf16.h>
#include <math.h>

// ---------------------------------------------------------------------------
// MoE block for MI455X (gfx1250, wave32): LN -> attn proj (+resid) ->
// positional-dispatch expert FFN (gelu) -> next linear (gelu).
// GEMMs: v_wmma_f32_16x16x32_bf16 with DOUBLE-BUFFERED async global->LDS
// staging (ASYNCcnt counted partial waits hide global latency behind WMMAs).
// Weights are converted fp32->bf16 AND transposed to [N][K] in a pre-pass so
// every LDS stage and fragment gather is contiguous (b128 only).
// ---------------------------------------------------------------------------

typedef __attribute__((ext_vector_type(16))) __bf16 v16bf;
typedef __attribute__((ext_vector_type(8)))  float  v8f;

#define LB_B    4
#define LB_S    4096
#define LB_D    1024
#define LB_E    8
#define LB_DFF  4096
#define LB_N    (LB_B * LB_S)           // 16384 tokens
#define LB_MB   2
#define LB_TPE  (LB_N / (LB_MB * LB_E)) // 1024 tokens / expert / micro-batch

static __device__ __forceinline__ unsigned short f2bf(float f) {
    unsigned int u = __float_as_uint(f);
    u += 0x7fffu + ((u >> 16) & 1u);     // round-to-nearest-even
    return (unsigned short)(u >> 16);
}

static __device__ __forceinline__ float gelu_exact(float v) {
    return 0.5f * v * (1.0f + erff(v * 0.70710678118654752f));
}

// async global -> LDS 128-bit copy (per-lane addresses), tracked by ASYNCcnt.
// Low 32 bits of a generic shared-memory address are the LDS byte address
// (flat aperture).
static __device__ __forceinline__ void async_b128(void* lds, const void* gptr) {
    unsigned           l = (unsigned)(unsigned long long)lds;
    unsigned long long g = (unsigned long long)gptr;
    asm volatile("global_load_async_to_lds_b128 %0, %1, off"
                 :: "v"(l), "v"(g) : "memory");
}
// ASYNCcnt completions are in-order: waiting <=6 guarantees every op except
// the newest 6 (the just-issued prefetch group) has retired.
static __device__ __forceinline__ void wait_async_le6() {
    asm volatile("s_wait_asynccnt 0x6" ::: "memory");
}
static __device__ __forceinline__ void wait_async0() {
    asm volatile("s_wait_asynccnt 0x0" ::: "memory");
}

// ------------------ fp32 -> bf16 convert WITH transpose --------------------
// in: [K][N] fp32 row-major; out: [N][K] bf16 row-major. 32x32 LDS tiles.
__global__ __launch_bounds__(256) void cvt_t_bf16(
    const float* __restrict__ in, unsigned short* __restrict__ out,
    int K, int N, long long gStride)
{
    in  += (long long)blockIdx.z * gStride;
    out += (long long)blockIdx.z * gStride;
    const int nb = blockIdx.x * 32;
    const int kb = blockIdx.y * 32;
    __shared__ float t[32][33];
    const int tx = threadIdx.x & 31;   // n within tile (load) / k (store)
    const int ty = threadIdx.x >> 5;   // 0..7
#pragma unroll
    for (int r = 0; r < 4; ++r)
        t[ty + 8 * r][tx] = in[(long long)(kb + ty + 8 * r) * N + nb + tx];
    __syncthreads();
#pragma unroll
    for (int r = 0; r < 4; ++r)
        out[(long long)(nb + ty + 8 * r) * K + kb + tx] = f2bf(t[tx][ty + 8 * r]);
}

// ------------------------- fused LayerNorm -> bf16 -------------------------
// one 256-thread block (8 wave32) per token row of D=1024
__global__ __launch_bounds__(256) void ln_to_bf16(
    const float* __restrict__ x, const float* __restrict__ g,
    const float* __restrict__ b, unsigned short* __restrict__ out)
{
    const int row = blockIdx.x;
    const int t   = threadIdx.x;
    const float4 xv = reinterpret_cast<const float4*>(x + (long long)row * LB_D)[t];

    __shared__ float r1[256];
    __shared__ float r2[256];
    r1[t] = xv.x + xv.y + xv.z + xv.w;
    r2[t] = xv.x * xv.x + xv.y * xv.y + xv.z * xv.z + xv.w * xv.w;
    __syncthreads();
#pragma unroll
    for (int st = 128; st > 0; st >>= 1) {
        if (t < st) { r1[t] += r1[t + st]; r2[t] += r2[t + st]; }
        __syncthreads();
    }
    const float mu  = r1[0] * (1.0f / LB_D);
    const float var = r2[0] * (1.0f / LB_D) - mu * mu;
    const float rs  = rsqrtf(var + 1e-5f);

    const int col = t * 4;
    const float4 gv = reinterpret_cast<const float4*>(g)[t];
    const float4 bv = reinterpret_cast<const float4*>(b)[t];
    unsigned short* o = out + (long long)row * LB_D + col;
    o[0] = f2bf((xv.x - mu) * rs * gv.x + bv.x);
    o[1] = f2bf((xv.y - mu) * rs * gv.y + bv.y);
    o[2] = f2bf((xv.z - mu) * rs * gv.z + bv.z);
    o[3] = f2bf((xv.w - mu) * rs * gv.w + bv.w);
}

// ------------------------------ WMMA GEMM ----------------------------------
// C[M,Nn] = A[M,K] @ B[K,Nn] where B is supplied TRANSPOSED: Bt[Nn][K].
// Block: 256 thr = 8 wave32, tile 128(M) x 64(N), K-step 64, double-buffered.
// Waves 4(M) x 2(N); each wave computes 32x32 => 4 f32 accumulators,
// 8 v_wmma_f32_16x16x32_bf16 per K-step (each fragment reused twice).
enum { EPI_RESID = 0, EPI_GELU_BF = 1, EPI_BIAS_BF = 2, EPI_GELU_F32 = 3 };

template <int EPI>
__global__ __launch_bounds__(256) void gemm_bf16_wmma(
    const unsigned short* __restrict__ A,     // [M,K] bf16
    const unsigned short* __restrict__ Bt,    // [Nn,K] bf16 (pre-transposed)
    const float* __restrict__ bias,           // [Nn]
    const float* __restrict__ resid,          // [M,Nn] fp32 (EPI_RESID only)
    unsigned short* __restrict__ outBf,       // bf16 out (chained stages)
    float* __restrict__ outF,                 // fp32 out (final stage)
    int M, int Nn, int K,
    long long aStride, long long bStride, long long biasStride, long long oStride,
    int groupMod)
{
    // per-group (micro-batch x expert) pointer adjust
    const int grp = blockIdx.z;
    const int e   = (groupMod > 0) ? (grp % groupMod) : 0;
    A    += (long long)grp * aStride;
    Bt   += (long long)e   * bStride;
    bias += (long long)e   * biasStride;
    if (EPI == EPI_GELU_F32) outF  += (long long)grp * oStride;
    else                     outBf += (long long)grp * oStride;

    const int tileM = blockIdx.y * 128;
    const int tileN = blockIdx.x * 64;

    // double-buffered tiles: [buf][row][k] (+8 pad keeps 16B alignment)
    __shared__ __align__(16) unsigned short At[2][128][72];
    __shared__ __align__(16) unsigned short Bs[2][64][72];

    const int tid  = threadIdx.x;
    const int lane = tid & 31;
    const int wid  = tid >> 5;
    const int wm   = wid & 3;    // wave's 32-row strip (0..3)
    const int wn   = wid >> 2;   // wave's 32-col strip (0..1)
    const int l16  = lane & 15;
    const int half = lane >> 4;  // 0: lanes 0-15, 1: lanes 16-31

    v8f acc[2][2] = {};

    // staging roles (contiguous b128 async copies; 6 async ops per wave/tile)
    const int ar = tid >> 1;            // A row 0..127
    const int ak = (tid & 1) * 32;      // A k-chunk 0/32
    const int br = tid >> 2;            // Bt row (n) 0..63
    const int bk = (tid & 3) * 16;      // Bt k-chunk 0/16/32/48

    auto issue_group = [&](int buf, int kb) {
        const unsigned short* Ag = A  + (long long)(tileM + ar) * K + kb + ak;
        const unsigned short* Bg = Bt + (long long)(tileN + br) * K + kb + bk;
#pragma unroll
        for (int u = 0; u < 4; ++u)
            async_b128(&At[buf][ar][ak + 8 * u], Ag + 8 * u);
#pragma unroll
        for (int u = 0; u < 2; ++u)
            async_b128(&Bs[buf][br][bk + 8 * u], Bg + 8 * u);
    };

    int cur = 0;
    issue_group(0, 0);                      // prologue: prefetch tile 0

    for (int kb = 0; kb < K; kb += 64) {
        const bool hasNext = (kb + 64) < K;
        if (hasNext) {
            issue_group(cur ^ 1, kb + 64);  // prefetch next tile into alt buf
            wait_async_le6();               // oldest group (cur tile) landed
        } else {
            wait_async0();                  // drain: last tile landed
        }
        __syncthreads();                    // cur tile visible to all waves

        // --- two 32-K sub-steps, 4 WMMAs each ---
#pragma unroll
        for (int s = 0; s < 2; ++s) {
            // A 16x32 frags: lane half h holds row m=l16, K = h*16 .. h*16+15
            v16bf a0, a1;
            {
                const int ko = s * 32 + half * 16;
                uint4* p = reinterpret_cast<uint4*>(&a0);
                const int r0 = wm * 32 + l16;
                p[0] = *reinterpret_cast<const uint4*>(&At[cur][r0][ko]);
                p[1] = *reinterpret_cast<const uint4*>(&At[cur][r0][ko + 8]);
                uint4* q = reinterpret_cast<uint4*>(&a1);
                const int r1 = r0 + 16;
                q[0] = *reinterpret_cast<const uint4*>(&At[cur][r1][ko]);
                q[1] = *reinterpret_cast<const uint4*>(&At[cur][r1][ko + 8]);
            }
            // B 32x16 frags: lane holds column n=l16, K = s*32 + half*16 ..+15
            v16bf b0, b1;
            {
                const int ko = s * 32 + half * 16;
                const int c0 = wn * 32 + l16;
                uint4* p = reinterpret_cast<uint4*>(&b0);
                p[0] = *reinterpret_cast<const uint4*>(&Bs[cur][c0][ko]);
                p[1] = *reinterpret_cast<const uint4*>(&Bs[cur][c0][ko + 8]);
                const int c1 = c0 + 16;
                uint4* q = reinterpret_cast<uint4*>(&b1);
                q[0] = *reinterpret_cast<const uint4*>(&Bs[cur][c1][ko]);
                q[1] = *reinterpret_cast<const uint4*>(&Bs[cur][c1][ko + 8]);
            }
            acc[0][0] = __builtin_amdgcn_wmma_f32_16x16x32_bf16(
                false, a0, false, b0, (short)0, acc[0][0], false, false);
            acc[0][1] = __builtin_amdgcn_wmma_f32_16x16x32_bf16(
                false, a0, false, b1, (short)0, acc[0][1], false, false);
            acc[1][0] = __builtin_amdgcn_wmma_f32_16x16x32_bf16(
                false, a1, false, b0, (short)0, acc[1][0], false, false);
            acc[1][1] = __builtin_amdgcn_wmma_f32_16x16x32_bf16(
                false, a1, false, b1, (short)0, acc[1][1], false, false);
        }

        __syncthreads();   // all waves done reading cur before it is refilled
        cur ^= 1;
    }

    // ------------------------------ epilogue -------------------------------
    // C/D layout: lane 0-15 -> N=l16, M = r; lane 16-31 -> M = r+8
#pragma unroll
    for (int i = 0; i < 2; ++i) {
#pragma unroll
        for (int j = 0; j < 2; ++j) {
            const int col = tileN + wn * 32 + j * 16 + l16;
            const float bcol = bias[col];
#pragma unroll
            for (int r = 0; r < 8; ++r) {
                const int row = tileM + wm * 32 + i * 16 + half * 8 + r;
                float v = acc[i][j][r] + bcol;
                const long long idx = (long long)row * Nn + col;
                if (EPI == EPI_RESID) {
                    v += resid[idx];
                    outBf[idx] = f2bf(v);
                } else if (EPI == EPI_GELU_BF) {
                    outBf[idx] = f2bf(gelu_exact(v));
                } else if (EPI == EPI_BIAS_BF) {
                    outBf[idx] = f2bf(v);
                } else {  // EPI_GELU_F32
                    outF[idx] = gelu_exact(v);
                }
            }
        }
    }
}

// ---------------------------------------------------------------------------
extern "C" void kernel_launch(void* const* d_in, const int* in_sizes, int n_in,
                              void* d_out, int out_size, void* d_ws, size_t ws_size,
                              hipStream_t stream) {
    (void)in_sizes; (void)n_in; (void)out_size; (void)ws_size;

    const float* x      = (const float*)d_in[0];
    const float* ln_g   = (const float*)d_in[1];
    const float* ln_b   = (const float*)d_in[2];
    const float* attn_w = (const float*)d_in[3];
    const float* attn_b = (const float*)d_in[4];
    // d_in[5] = gate_w : dead code in the reference (top_k result unused)
    const float* w1     = (const float*)d_in[6];
    const float* b1     = (const float*)d_in[7];
    const float* w2     = (const float*)d_in[8];
    const float* b2     = (const float*)d_in[9];
    const float* next_w = (const float*)d_in[10];
    const float* next_b = (const float*)d_in[11];
    float* out          = (float*)d_out;

    // ---- workspace carve (bf16 buffers; weights stored transposed) ----
    char* ws = (char*)d_ws;
    size_t off = 0;
    auto carve = [&](size_t bytes) {
        char* p = ws + off;
        off += (bytes + 255) & ~size_t(255);
        return p;
    };
    unsigned short* attn_wt = (unsigned short*)carve((size_t)LB_D * LB_D * 2);
    unsigned short* next_wt = (unsigned short*)carve((size_t)LB_D * LB_D * 2);
    unsigned short* w1t     = (unsigned short*)carve((size_t)LB_E * LB_D * LB_DFF * 2);
    unsigned short* w2t     = (unsigned short*)carve((size_t)LB_E * LB_DFF * LB_D * 2);
    unsigned short* xn      = (unsigned short*)carve((size_t)LB_N * LB_D * 2);
    unsigned short* xattn   = (unsigned short*)carve((size_t)LB_N * LB_D * 2);
    unsigned short* hbuf    = (unsigned short*)carve((size_t)LB_N * LB_DFF * 2);
    unsigned short* ybuf    = (unsigned short*)carve((size_t)LB_N * LB_D * 2);

    // ---- 1) weights -> bf16, transposed to [N][K] ----
    cvt_t_bf16<<<dim3(LB_D / 32, LB_D / 32, 1), dim3(256), 0, stream>>>(
        attn_w, attn_wt, LB_D, LB_D, 0);
    cvt_t_bf16<<<dim3(LB_D / 32, LB_D / 32, 1), dim3(256), 0, stream>>>(
        next_w, next_wt, LB_D, LB_D, 0);
    cvt_t_bf16<<<dim3(LB_DFF / 32, LB_D / 32, LB_E), dim3(256), 0, stream>>>(
        w1, w1t, LB_D, LB_DFF, (long long)LB_D * LB_DFF);
    cvt_t_bf16<<<dim3(LB_D / 32, LB_DFF / 32, LB_E), dim3(256), 0, stream>>>(
        w2, w2t, LB_DFF, LB_D, (long long)LB_DFF * LB_D);

    // ---- 2) fused LayerNorm -> bf16 ----
    ln_to_bf16<<<dim3(LB_N), dim3(256), 0, stream>>>(x, ln_g, ln_b, xn);

    // ---- 3) x_attn = ln(x) @ attn_w + attn_b + x   (bf16 out) ----
    gemm_bf16_wmma<EPI_RESID><<<dim3(LB_D / 64, LB_N / 128, 1), dim3(256), 0, stream>>>(
        xn, attn_wt, attn_b, x, xattn, nullptr,
        LB_N, LB_D, LB_D, 0, 0, 0, 0, 0);

    // ---- 4) h = gelu(x_attn[g] @ w1[e] + b1[e])  per (micro-batch, expert) ----
    gemm_bf16_wmma<EPI_GELU_BF><<<dim3(LB_DFF / 64, LB_TPE / 128, LB_MB * LB_E), dim3(256), 0, stream>>>(
        xattn, w1t, b1, nullptr, hbuf, nullptr,
        LB_TPE, LB_DFF, LB_D,
        (long long)LB_TPE * LB_D,            // A group stride
        (long long)LB_D * LB_DFF,            // B expert stride
        (long long)LB_DFF,                   // bias expert stride
        (long long)LB_TPE * LB_DFF,          // out group stride
        LB_E);

    // ---- 5) y = h[g] @ w2[e] + b2[e] ----
    gemm_bf16_wmma<EPI_BIAS_BF><<<dim3(LB_D / 64, LB_TPE / 128, LB_MB * LB_E), dim3(256), 0, stream>>>(
        hbuf, w2t, b2, nullptr, ybuf, nullptr,
        LB_TPE, LB_D, LB_DFF,
        (long long)LB_TPE * LB_DFF,
        (long long)LB_DFF * LB_D,
        (long long)LB_D,
        (long long)LB_TPE * LB_D,
        LB_E);

    // ---- 6) out = gelu(y @ next_w + next_b)  (fp32 out) ----
    gemm_bf16_wmma<EPI_GELU_F32><<<dim3(LB_D / 64, LB_N / 128, 1), dim3(256), 0, stream>>>(
        ybuf, next_wt, next_b, nullptr, nullptr, out,
        LB_N, LB_D, LB_D, 0, 0, 0, 0, 0);
}
